// Attention_40681930227760
// MI455X (gfx1250) — matrix operand order repaired and checked by reading, not compiled, not run on hardware
//
#include <hip/hip_runtime.h>

// ---------------------------------------------------------------------------
// Types for CDNA5 WMMA (gfx1250, wave32)
// ---------------------------------------------------------------------------
typedef __bf16 bf16_t;
typedef bf16_t v16bf __attribute__((ext_vector_type(16)));
typedef float  v8f   __attribute__((ext_vector_type(8)));

typedef __attribute__((address_space(3))) unsigned short lds_ushort_t;

union Frag16 {
    v16bf          bf;
    unsigned int   u32[8];
    unsigned short u16[16];
};

static __device__ __forceinline__ unsigned short f2bf(float f) {
    unsigned int u = __builtin_bit_cast(unsigned int, f);
    u += 0x7FFFu + ((u >> 16) & 1u);          // round-to-nearest-even
    return (unsigned short)(u >> 16);
}

// A-matrix fragment (16x32 bf16, M x K), row-major source, contiguous k order used by this code:
//   lane: M = lane&15 ; element e=(v,h): K = 16*(lane>=16) + 2*v + h
static __device__ __forceinline__ void load_frag_A(const unsigned short* p, int stride,
                                                   int lane, Frag16& f) {
    const unsigned short* row = p + (lane & 15) * stride;
    const int kb = (lane & 16) ? 16 : 0;
#pragma unroll
    for (int v = 0; v < 8; ++v) {
        int k = kb + 2 * v;
        f.u32[v] = *(const unsigned int*)(row + k);    // K pairs contiguous
    }
}

// B-matrix fragment (32x16 bf16, K x N), source stored N-major (column of B
// contiguous in K): lane: N = lane&15 ; K = 16*(lane>=16) + 2*v + h
static __device__ __forceinline__ void load_frag_Bc(const unsigned short* p, int stride,
                                                    int lane, Frag16& f) {
    const unsigned short* col = p + (lane & 15) * stride;
    const int kb = (lane & 16) ? 16 : 0;
#pragma unroll
    for (int v = 0; v < 8; ++v)
        f.u32[v] = *(const unsigned int*)(col + kb + 2 * v);
}

// B-matrix fragment where source is K-major with row stride `kstride` (scalar loads)
static __device__ __forceinline__ void load_frag_Bs(const unsigned short* p, int kstride,
                                                    int lane, Frag16& f) {
    const int n  = lane & 15;
    const int kb = (lane & 16) ? 16 : 0;
#pragma unroll
    for (int e = 0; e < 16; ++e) {
        int v = e >> 1, h = e & 1;
        int k = kb + 2 * v + h;
        f.u16[e] = p[k * kstride + n];
    }
}

static __device__ __forceinline__ v8f wmma_bf16(const Frag16& a, const Frag16& b, v8f c) {
    return __builtin_amdgcn_wmma_f32_16x16x32_bf16(false, a.bf, false, b.bf,
                                                   (short)0, c, false, false);
}

// ---------------------------------------------------------------------------
// Problem constants
// ---------------------------------------------------------------------------
#define SEQ 4096
#define DM  768
#define NH  12
#define HD  64

// ---------------------------------------------------------------------------
// Conversion kernels
// ---------------------------------------------------------------------------
__global__ void cvt_bf16(const float* __restrict__ in, unsigned short* __restrict__ out, int n) {
    int i = blockIdx.x * blockDim.x + threadIdx.x;
    if (i < n) out[i] = f2bf(in[i]);
}

// in[rows][cols] (fp32) -> out[cols][rows] (bf16)  (weights stored K-contiguous)
__global__ void cvt_transpose(const float* __restrict__ in, unsigned short* __restrict__ out,
                              int rows, int cols) {
    int i = blockIdx.x * blockDim.x + threadIdx.x;
    if (i < rows * cols) {
        int r = i / cols, c = i % cols;
        out[c * rows + r] = f2bf(in[i]);
    }
}

// ---------------------------------------------------------------------------
// Kernel 1: QKV GEMM (bf16 WMMA, fp32 acc) + per-head LayerNorm on q,k
//   grid (64 row-tiles, 36 = which*12+head), 128 threads = 4 waves,
//   each wave: 16 rows x 64 cols, K = 768 in 24 WMMA steps.
// ---------------------------------------------------------------------------
__global__ __launch_bounds__(128) void qkv_ln_kernel(
    const unsigned short* __restrict__ xb,      // [4096][768] bf16
    const unsigned short* __restrict__ wqkvT,   // [2304][768] bf16
    const float* __restrict__ qg, const float* __restrict__ qbeta,
    const float* __restrict__ kg, const float* __restrict__ kbeta,
    unsigned short* __restrict__ qh,            // [12][4096][64] bf16 (scaled)
    unsigned short* __restrict__ kh,
    unsigned short* __restrict__ vh)
{
    const int lane    = threadIdx.x & 31;
    const int wave    = threadIdx.x >> 5;
    const int rowbase = blockIdx.x * 64 + wave * 16;
    const int which   = blockIdx.y / NH;
    const int head    = blockIdx.y % NH;
    const int colbase = which * DM + head * HD;

    v8f acc[4];
#pragma unroll
    for (int t = 0; t < 4; ++t)
#pragma unroll
        for (int r = 0; r < 8; ++r) acc[t][r] = 0.0f;

    for (int kk = 0; kk < DM; kk += 32) {
        Frag16 A;
        load_frag_A(xb + rowbase * DM + kk, DM, lane, A);
#pragma unroll
        for (int t = 0; t < 4; ++t) {
            Frag16 B;
            load_frag_Bc(wqkvT + (colbase + t * 16) * DM + kk, DM, lane, B);
            acc[t] = wmma_bf16(A, B, acc[t]);
        }
    }

    // per-row LayerNorm statistics over 64 cols (rows distributed per C layout)
    float mean[8], rstd[8];
    if (which != 2) {
#pragma unroll
        for (int r = 0; r < 8; ++r) {
            float s1 = 0.f, s2 = 0.f;
#pragma unroll
            for (int t = 0; t < 4; ++t) { float v = acc[t][r]; s1 += v; s2 += v * v; }
#pragma unroll
            for (int m = 1; m < 16; m <<= 1) {
                s1 += __shfl_xor(s1, m, 32);
                s2 += __shfl_xor(s2, m, 32);
            }
            float mu  = s1 * (1.0f / HD);
            float var = s2 * (1.0f / HD) - mu * mu;
            mean[r] = mu;
            rstd[r] = rsqrtf(var + 1e-3f);
        }
    }

    const float* gp = (which == 0) ? qg : kg;
    const float* bp = (which == 0) ? qbeta : kbeta;
    float gv[4], bv[4];
#pragma unroll
    for (int t = 0; t < 4; ++t) {
        int ch = t * 16 + (lane & 15);
        gv[t] = (which != 2) ? gp[ch] : 1.0f;
        bv[t] = (which != 2) ? bp[ch] : 0.0f;
    }
    const float smul = (which == 0) ? 0.125f : 1.0f;   // hd^-0.5 folded into q
    unsigned short* dst = (which == 0) ? qh : (which == 1) ? kh : vh;

#pragma unroll
    for (int r = 0; r < 8; ++r) {
        int row = r + ((lane & 16) ? 8 : 0);
#pragma unroll
        for (int t = 0; t < 4; ++t) {
            float v = acc[t][r];
            if (which != 2) v = ((v - mean[r]) * rstd[r] * gv[t] + bv[t]) * smul;
            int ch = t * 16 + (lane & 15);
            dst[(head * SEQ + rowbase + row) * HD + ch] = f2bf(v);
        }
    }
}

// ---------------------------------------------------------------------------
// Kernel 2: flash attention. grid (64 q-tiles, 12 heads), 128 threads = 4 waves,
//   each wave: 16 query rows x hd=64, streaming 32-key blocks.
//   K/V tiles (4KB each, contiguous in head-major layout) are staged into LDS
//   ONCE per workgroup via CDNA5 async-to-LDS loads (ASYNCcnt path) and shared
//   by all 4 waves; fragments are then built from LDS.
// ---------------------------------------------------------------------------
__global__ __launch_bounds__(128) void flash_kernel(
    const unsigned short* __restrict__ qh,
    const unsigned short* __restrict__ kh,
    const unsigned short* __restrict__ vh,
    unsigned short* __restrict__ ob)            // [4096][768] bf16
{
    __shared__ __align__(16) unsigned short lds_k[32][HD];   // 4KB: K tile [key][ch]
    __shared__ __align__(16) unsigned short lds_v[32][HD];   // 4KB: V tile [key][ch]
    __shared__ __align__(16) unsigned short lds_p[4][16][32];

    const int tid   = threadIdx.x;
    const int lane  = tid & 31;
    const int wave  = tid >> 5;
    const int qbase = blockIdx.x * 64 + wave * 16;
    const int head  = blockIdx.y;

    // raw LDS byte offsets for the async-load destinations
    const unsigned kOff = (unsigned)(unsigned long long)(lds_ushort_t*)&lds_k[0][0];
    const unsigned vOff = (unsigned)(unsigned long long)(lds_ushort_t*)&lds_v[0][0];

    Frag16 aq[2];
#pragma unroll
    for (int kh2 = 0; kh2 < 2; ++kh2)
        load_frag_A(qh + (head * SEQ + qbase) * HD + kh2 * 32, HD, lane, aq[kh2]);

    v8f o[4];
    float mrow[8], lrow[8];
#pragma unroll
    for (int t = 0; t < 4; ++t)
#pragma unroll
        for (int r = 0; r < 8; ++r) o[t][r] = 0.0f;
#pragma unroll
    for (int r = 0; r < 8; ++r) { mrow[r] = -1e30f; lrow[r] = 0.0f; }

    for (int kb = 0; kb < SEQ; kb += 32) {
        __syncthreads();   // all waves done reading previous K/V tile

        // stage K and V tiles (2 x 4KB contiguous) -> LDS, 4 x b128 per thread
        {
            const unsigned long long kSrc =
                (unsigned long long)(kh + (head * SEQ + kb) * HD);
            const unsigned long long vSrc =
                (unsigned long long)(vh + (head * SEQ + kb) * HD);
            unsigned co = (unsigned)tid * 16u;     // 128 threads x 16B = 2KB/sweep
#pragma unroll
            for (int it = 0; it < 2; ++it) {
                asm volatile("global_load_async_to_lds_b128 %0, %1, off"
                             :: "v"(kOff + co), "v"(kSrc + co) : "memory");
                asm volatile("global_load_async_to_lds_b128 %0, %1, off"
                             :: "v"(vOff + co), "v"(vSrc + co) : "memory");
                co += 2048u;
            }
        }
        asm volatile("s_wait_asynccnt 0" ::: "memory");
        __syncthreads();   // everyone's tile data visible

        // S = (scaled,normed Q) @ K^T : 16 x 32 keys, 2 C tiles, K(hd)=64
        v8f sacc[2];
#pragma unroll
        for (int t = 0; t < 2; ++t)
#pragma unroll
            for (int r = 0; r < 8; ++r) sacc[t][r] = 0.0f;
#pragma unroll
        for (int kh2 = 0; kh2 < 2; ++kh2)
#pragma unroll
            for (int t = 0; t < 2; ++t) {
                Frag16 B;
                load_frag_Bc(&lds_k[t * 16][kh2 * 32], HD, lane, B);
                sacc[t] = wmma_bf16(aq[kh2], B, sacc[t]);
            }

        // online softmax per row
#pragma unroll
        for (int r = 0; r < 8; ++r) {
            float mx = fmaxf(sacc[0][r], sacc[1][r]);
#pragma unroll
            for (int m = 1; m < 16; m <<= 1) mx = fmaxf(mx, __shfl_xor(mx, m, 32));
            float mn    = fmaxf(mrow[r], mx);
            float alpha = expf(mrow[r] - mn);
            int   row   = r + ((lane & 16) ? 8 : 0);
            float ps    = 0.0f;
#pragma unroll
            for (int t = 0; t < 2; ++t) {
                float p = expf(sacc[t][r] - mn);
                ps += p;
                lds_p[wave][row][t * 16 + (lane & 15)] = f2bf(p);
            }
#pragma unroll
            for (int m = 1; m < 16; m <<= 1) ps += __shfl_xor(ps, m, 32);
            lrow[r] = lrow[r] * alpha + ps;
            mrow[r] = mn;
#pragma unroll
            for (int t = 0; t < 4; ++t) o[t][r] *= alpha;
        }

        // O += P @ V : P staged via LDS (C-layout -> A-layout), V from LDS tile
        Frag16 Ap;
        load_frag_A(&lds_p[wave][0][0], 32, lane, Ap);
#pragma unroll
        for (int t = 0; t < 4; ++t) {
            Frag16 Bv;
            load_frag_Bs(&lds_v[0][t * 16], HD, lane, Bv);
            o[t] = wmma_bf16(Ap, Bv, o[t]);
        }
    }

    // normalize and emit bf16 in [S][D] layout for the projection GEMM
#pragma unroll
    for (int r = 0; r < 8; ++r) {
        float inv = 1.0f / lrow[r];
        int   row = r + ((lane & 16) ? 8 : 0);
#pragma unroll
        for (int t = 0; t < 4; ++t)
            ob[(qbase + row) * DM + head * HD + t * 16 + (lane & 15)] = f2bf(o[t][r] * inv);
    }
}

// ---------------------------------------------------------------------------
// Kernel 3: output projection  out = ob @ Wproj + b  (fp32 out)
// ---------------------------------------------------------------------------
__global__ __launch_bounds__(128) void proj_kernel(
    const unsigned short* __restrict__ ob,
    const unsigned short* __restrict__ wprojT,  // [768][768] bf16, col-major
    const float* __restrict__ bias,
    float* __restrict__ out)
{
    const int lane    = threadIdx.x & 31;
    const int wave    = threadIdx.x >> 5;
    const int rowbase = blockIdx.x * 64 + wave * 16;
    const int colbase = blockIdx.y * 64;

    v8f acc[4];
#pragma unroll
    for (int t = 0; t < 4; ++t)
#pragma unroll
        for (int r = 0; r < 8; ++r) acc[t][r] = 0.0f;

    for (int kk = 0; kk < DM; kk += 32) {
        Frag16 A;
        load_frag_A(ob + rowbase * DM + kk, DM, lane, A);
#pragma unroll
        for (int t = 0; t < 4; ++t) {
            Frag16 B;
            load_frag_Bc(wprojT + (colbase + t * 16) * DM + kk, DM, lane, B);
            acc[t] = wmma_bf16(A, B, acc[t]);
        }
    }

#pragma unroll
    for (int t = 0; t < 4; ++t) {
        int   col  = colbase + t * 16 + (lane & 15);
        float bval = bias[col];
#pragma unroll
        for (int r = 0; r < 8; ++r) {
            int row = r + ((lane & 16) ? 8 : 0);
            out[(rowbase + row) * DM + col] = acc[t][r] + bval;
        }
    }
}

// ---------------------------------------------------------------------------
// Launch
// ---------------------------------------------------------------------------
extern "C" void kernel_launch(void* const* d_in, const int* in_sizes, int n_in,
                              void* d_out, int out_size, void* d_ws, size_t ws_size,
                              hipStream_t stream) {
    const float* x     = (const float*)d_in[0];
    const float* wqkv  = (const float*)d_in[1];
    const float* qg    = (const float*)d_in[2];
    const float* qb    = (const float*)d_in[3];
    const float* kg    = (const float*)d_in[4];
    const float* kbeta = (const float*)d_in[5];
    const float* wproj = (const float*)d_in[6];
    const float* bproj = (const float*)d_in[7];
    float* out = (float*)d_out;

    char* ws = (char*)d_ws;
    unsigned short* xb     = (unsigned short*)(ws);                     // 4096*768
    unsigned short* wqkvT  = (unsigned short*)(ws + 6291456);           // 2304*768
    unsigned short* wprojT = (unsigned short*)(ws + 9830400);           // 768*768
    unsigned short* qh     = (unsigned short*)(ws + 11010048);          // 12*4096*64
    unsigned short* kh     = (unsigned short*)(ws + 17301504);
    unsigned short* vh     = (unsigned short*)(ws + 23592960);
    unsigned short* ob     = (unsigned short*)(ws + 29884416);          // 4096*768

    const int nx = SEQ * DM;
    cvt_bf16<<<(nx + 255) / 256, 256, 0, stream>>>(x, xb, nx);
    cvt_transpose<<<(DM * 3 * DM + 255) / 256, 256, 0, stream>>>(wqkv, wqkvT, DM, 3 * DM);
    cvt_transpose<<<(DM * DM + 255) / 256, 256, 0, stream>>>(wproj, wprojT, DM, DM);

    qkv_ln_kernel<<<dim3(SEQ / 64, 3 * NH), 128, 0, stream>>>(
        xb, wqkvT, qg, qb, kg, kbeta, qh, kh, vh);

    flash_kernel<<<dim3(SEQ / 64, NH), 128, 0, stream>>>(qh, kh, vh, ob);

    proj_kernel<<<dim3(SEQ / 64, NH), 128, 0, stream>>>(ob, wprojT, bproj, out);
}
